// GroupedLinearRef_18356690223822
// MI455X (gfx1250) — compile-verified, matrix operand order checked by reading, not run on hardware
//
#include <hip/hip_runtime.h>

#define T_TOK 16384
#define IN_F  1024
#define OUT_F 1024
#define NEXP  8

#define BM 128
#define BN 128
#define BK 32
#define KS 40                 // padded K stride (bf16 elements): 80B rows -> 16B aligned
#define NK (IN_F / BK)        // 32 k-steps

typedef __attribute__((ext_vector_type(16))) __bf16 v16bf;
typedef __attribute__((ext_vector_type(8)))  float  v8f;
typedef __attribute__((ext_vector_type(4)))  unsigned short us4;
typedef __attribute__((ext_vector_type(8)))  unsigned short us8;
typedef __attribute__((ext_vector_type(16))) unsigned short us16;

__device__ __forceinline__ unsigned short f2bf_rne(float f) {
    unsigned u = __builtin_bit_cast(unsigned, f);
    u += 0x7FFFu + ((u >> 16) & 1u);          // round-to-nearest-even
    return (unsigned short)(u >> 16);
}
__device__ __forceinline__ float bf2f(unsigned short h) {
    unsigned u = ((unsigned)h) << 16;
    return __builtin_bit_cast(float, u);
}

__global__ __launch_bounds__(256) void grouped_gemm_bf16x3(
    const float* __restrict__ x, const float* __restrict__ w,
    const int* __restrict__ seg_lens, float* __restrict__ y)
{
    __shared__ __align__(16) unsigned short sAhi[BM * KS];
    __shared__ __align__(16) unsigned short sAlo[BM * KS];
    __shared__ __align__(16) unsigned short sBhi[BN * KS];
    __shared__ __align__(16) unsigned short sBlo[BN * KS];

    const int tid   = threadIdx.x;
    const int lane  = tid & 31;
    const int wid   = tid >> 5;
    const int waveM = wid >> 2;              // 0..1
    const int waveN = wid & 3;               // 0..3
    const int mBase = waveM * 64;
    const int nBase = waveN * 32;
    const int lrow  = lane & 15;             // row/col within 16x16 fragment
    const int kh    = (lane >> 4) << 3;      // 0 or 8: K-half per WMMA layout
    const int col0  = blockIdx.x * BN;
    const int row0  = blockIdx.y * BM;

    // cumulative segment bounds (uniform scalar loads)
    int bounds[NEXP];
    int run = 0;
#pragma unroll
    for (int e = 0; e < NEXP; ++e) { run += seg_lens[e]; bounds[e] = run; }

    // cooperative tile-loader indexing: 256 threads, 32 floats (8 float4) per row
    const int lr = tid >> 3;                 // base row 0..31 (stride 32 over 128)
    const int lc = (tid & 7) << 2;           // float column 0,4,...,28

    for (int e = 0; e < NEXP; ++e) {
        const int segLo = (e == 0) ? 0 : bounds[e - 1];
        const int segHi = bounds[e];
        const int rs = max(row0, segLo);
        const int re = min(row0 + BM, segHi);
        if (rs >= re) continue;              // expert does not touch this row stripe

        const float* __restrict__ we = w + (size_t)e * OUT_F * IN_F;

        v8f acc[4][2] = {};

        float4 aReg[4], bReg[4];
#pragma unroll
        for (int p = 0; p < 4; ++p) {        // prologue: k-tile 0
            const int r = lr + p * 32;
            aReg[p] = *(const float4*)(x  + (size_t)(row0 + r) * IN_F + lc);
            bReg[p] = *(const float4*)(we + (size_t)(col0 + r) * IN_F + lc);
        }

        for (int kt = 0; kt < NK; ++kt) {
            // split f32 -> bf16 hi/lo planes in LDS
#pragma unroll
            for (int p = 0; p < 4; ++p) {
                const int r = lr + p * 32;
                float av[4] = {aReg[p].x, aReg[p].y, aReg[p].z, aReg[p].w};
                float bv[4] = {bReg[p].x, bReg[p].y, bReg[p].z, bReg[p].w};
                us4 ahi, alo, bhi, blo;
#pragma unroll
                for (int q = 0; q < 4; ++q) {
                    unsigned short h = f2bf_rne(av[q]);
                    ahi[q] = h;
                    alo[q] = f2bf_rne(av[q] - bf2f(h));
                    unsigned short g = f2bf_rne(bv[q]);
                    bhi[q] = g;
                    blo[q] = f2bf_rne(bv[q] - bf2f(g));
                }
                *(us4*)(&sAhi[r * KS + lc]) = ahi;
                *(us4*)(&sAlo[r * KS + lc]) = alo;
                *(us4*)(&sBhi[r * KS + lc]) = bhi;
                *(us4*)(&sBlo[r * KS + lc]) = blo;
            }
            // register prefetch of next k-tile (overlaps WMMA phase)
            if (kt + 1 < NK) {
                const int k0 = (kt + 1) * BK;
#pragma unroll
                for (int p = 0; p < 4; ++p) {
                    const int r = lr + p * 32;
                    aReg[p] = *(const float4*)(x  + (size_t)(row0 + r) * IN_F + k0 + lc);
                    bReg[p] = *(const float4*)(we + (size_t)(col0 + r) * IN_F + k0 + lc);
                }
            }
            __syncthreads();                 // LDS tiles visible

            // fragment loads: two contiguous 16B ds reads per fragment
            v16bf aHi[4], aLo[4], bHi[2], bLo[2];
#pragma unroll
            for (int i = 0; i < 4; ++i) {
                const int rr = mBase + i * 16 + lrow;
                us16 t;
                ((us8*)&t)[0] = *(const us8*)(&sAhi[rr * KS + kh]);
                ((us8*)&t)[1] = *(const us8*)(&sAhi[rr * KS + kh + 16]);
                aHi[i] = __builtin_bit_cast(v16bf, t);
                ((us8*)&t)[0] = *(const us8*)(&sAlo[rr * KS + kh]);
                ((us8*)&t)[1] = *(const us8*)(&sAlo[rr * KS + kh + 16]);
                aLo[i] = __builtin_bit_cast(v16bf, t);
            }
#pragma unroll
            for (int j = 0; j < 2; ++j) {
                const int cc = nBase + j * 16 + lrow;
                us16 t;
                ((us8*)&t)[0] = *(const us8*)(&sBhi[cc * KS + kh]);
                ((us8*)&t)[1] = *(const us8*)(&sBhi[cc * KS + kh + 16]);
                bHi[j] = __builtin_bit_cast(v16bf, t);
                ((us8*)&t)[0] = *(const us8*)(&sBlo[cc * KS + kh]);
                ((us8*)&t)[1] = *(const us8*)(&sBlo[cc * KS + kh + 16]);
                bLo[j] = __builtin_bit_cast(v16bf, t);
            }

            // split-bf16 fp32 emulation: hi*hi + hi*lo + lo*hi
#pragma unroll
            for (int i = 0; i < 4; ++i)
#pragma unroll
                for (int j = 0; j < 2; ++j) {
                    acc[i][j] = __builtin_amdgcn_wmma_f32_16x16x32_bf16(
                        false, aHi[i], false, bHi[j], (short)0, acc[i][j], false, false);
                    acc[i][j] = __builtin_amdgcn_wmma_f32_16x16x32_bf16(
                        false, aHi[i], false, bLo[j], (short)0, acc[i][j], false, false);
                    acc[i][j] = __builtin_amdgcn_wmma_f32_16x16x32_bf16(
                        false, aLo[i], false, bHi[j], (short)0, acc[i][j], false, false);
                }

            __syncthreads();                 // all reads done before next overwrite
        }

        // store: C/D layout -> VGPR r holds M=r (lanes 0-15) / M=r+8 (lanes 16-31)
#pragma unroll
        for (int i = 0; i < 4; ++i) {
            const int rbase = row0 + mBase + i * 16 + ((lane >> 4) << 3);
#pragma unroll
            for (int j = 0; j < 2; ++j) {
                const int col = col0 + nBase + j * 16 + (lane & 15);
#pragma unroll
                for (int r = 0; r < 8; ++r) {
                    const int row = rbase + r;
                    if (row >= rs && row < re)
                        y[(size_t)row * OUT_F + col] = acc[i][j][r];
                }
            }
        }
    }
}

extern "C" void kernel_launch(void* const* d_in, const int* in_sizes, int n_in,
                              void* d_out, int out_size, void* d_ws, size_t ws_size,
                              hipStream_t stream) {
    const float* x   = (const float*)d_in[0];
    const float* w   = (const float*)d_in[1];
    const int*   seg = (const int*)d_in[2];
    float*       y   = (float*)d_out;
    (void)in_sizes; (void)n_in; (void)out_size; (void)d_ws; (void)ws_size;

    dim3 grid(OUT_F / BN, T_TOK / BM);       // 8 x 128 blocks
    dim3 block(256);                          // 8 wave32s
    hipLaunchKernelGGL(grouped_gemm_bf16x3, grid, block, 0, stream, x, w, seg, y);
}
